// OneHopRanker_69020124447184
// MI455X (gfx1250) — matrix operand order repaired and checked by reading, not compiled, not run on hardware
//
#include <hip/hip_runtime.h>
#include <hip/hip_bf16.h>

// ---------------- problem constants (fixed by the reference) ----------------
#define NCQ   256     // number of query contexts
#define DIME  128     // embedding dim
#define KSEL  32      // top-k width
#define ROWS  16      // queries per block (one WMMA M tile)
#define CHUNK 2048    // candidates per block
#define LDSTRIDE (CHUNK + 4)   // pad to break bank alignment

typedef __attribute__((ext_vector_type(16))) __bf16 v16bf;
typedef __attribute__((ext_vector_type(8)))  float  v8f;

union BFrag {
    v16bf v;
    __bf16 e[16];
    unsigned short u[16];
};

// ------------------------- wave32 top-k primitives --------------------------
// One (value,index) per lane; list kept sorted descending across lanes 0..31.

__device__ __forceinline__ void cmpswap(float& v, int& i, int lane, int j, bool dir) {
    float pv = __shfl_xor(v, j);
    int   pi = __shfl_xor(i, j);
    bool lower   = (lane & j) == 0;
    bool takeMax = (lower == dir);             // dir==true => descending block
    bool take    = takeMax ? (pv > v) : (pv < v);
    if (take) { v = pv; i = pi; }
}

__device__ __forceinline__ void bitonic_sort32_desc(float& v, int& i, int lane) {
    #pragma unroll
    for (int k = 2; k <= 32; k <<= 1) {
        bool dir = ((lane & k) == 0);
        #pragma unroll
        for (int j = k >> 1; j >= 1; j >>= 1) cmpswap(v, i, lane, j, dir);
    }
}

// merge a sorted-descending candidate list (nv,ni) into the best list (bv,bi)
__device__ __forceinline__ void merge32_desc(float& bv, int& bi, float nv, int ni, int lane) {
    float rv = __shfl_xor(nv, 31);             // reverse -> ascending
    int   ri = __shfl_xor(ni, 31);
    if (rv > bv) { bv = rv; bi = ri; }         // elementwise max -> bitonic top-32
    #pragma unroll
    for (int j = 16; j >= 1; j >>= 1) cmpswap(bv, bi, lane, j, true);
}

// ------------------------------- kernels ------------------------------------

__global__ void cvt_bf16_kernel(const float* __restrict__ src, __bf16* __restrict__ dst, int n) {
    int i = blockIdx.x * blockDim.x + threadIdx.x;
    if (i < n) dst[i] = (__bf16)src[i];
}

// Score ROWS queries against CHUNK candidates with bf16 WMMA, then exact
// per-query top-32 within the chunk (wave-select over LDS-staged scores).
// blockIdx.x = query tile (NCQ/ROWS), blockIdx.y = candidate chunk.
__global__ __launch_bounds__(256)
void score_topk_kernel(const __bf16* __restrict__ A,      // [NCQ, DIME] bf16
                       const float*  __restrict__ bank,   // [N, DIME] f32
                       float* __restrict__ pV, int* __restrict__ pI,
                       int N, int nChunks)
{
    __shared__ float sims[ROWS][LDSTRIDE];

    const int qt   = blockIdx.x;
    const int chunk= blockIdx.y;
    const int tid  = threadIdx.x;
    const int lane = tid & 31;
    const int wave = tid >> 5;
    const int c0   = chunk * CHUNK;
    const int m    = lane & 15;      // M row (A) / N col (B) within tile
    const int hi   = lane >> 4;      // lane half selects K sub-range

    // ---- A fragments: 4 K-steps of 16x32 bf16 (contiguous K: half h holds K 16h..16h+15) ----
    BFrag afrag[4];
    {
        const unsigned short* arow = (const unsigned short*)(A + (qt * ROWS + m) * DIME);
        #pragma unroll
        for (int s = 0; s < 4; ++s) {
            int kb = s * 32 + hi * 16;
            #pragma unroll
            for (int j = 0; j < 8; ++j) {
                afrag[s].u[j]     = arow[kb + j];        // K = kb .. kb+7
                afrag[s].u[8 + j] = arow[kb + 8 + j];    // K = kb+8 .. kb+15
            }
        }
    }

    // ---- score: each wave owns every 8th 16-candidate subtile ----
    for (int sub = wave; sub < CHUNK / 16; sub += 8) {
        int cand  = c0 + sub * 16 + m;
        int candc = cand < N ? cand : N - 1;             // clamp; masked later
        const float* brow = bank + (long)candc * DIME;
        v8f acc = {};
        #pragma unroll
        for (int s = 0; s < 4; ++s) {
            int kb = s * 32 + hi * 16;                   // 16 contiguous K per lane
            const float4* p = (const float4*)(brow + kb);
            float4 x0 = p[0], x1 = p[1], x2 = p[2], x3 = p[3];
            BFrag bf;
            bf.e[0]=(__bf16)x0.x;  bf.e[1]=(__bf16)x0.y;  bf.e[2]=(__bf16)x0.z;  bf.e[3]=(__bf16)x0.w;
            bf.e[4]=(__bf16)x1.x;  bf.e[5]=(__bf16)x1.y;  bf.e[6]=(__bf16)x1.z;  bf.e[7]=(__bf16)x1.w;
            bf.e[8]=(__bf16)x2.x;  bf.e[9]=(__bf16)x2.y;  bf.e[10]=(__bf16)x2.z; bf.e[11]=(__bf16)x2.w;
            bf.e[12]=(__bf16)x3.x; bf.e[13]=(__bf16)x3.y; bf.e[14]=(__bf16)x3.z; bf.e[15]=(__bf16)x3.w;
            acc = __builtin_amdgcn_wmma_f32_16x16x32_bf16(
                      false, afrag[s].v, false, bf.v, (short)0, acc, false, false);
        }
        // C layout: vgpr r, lane -> (M = r + 8*hi, N = lane&15)
        #pragma unroll
        for (int r = 0; r < 8; ++r)
            sims[r + hi * 8][sub * 16 + m] = acc[r];
    }

    __syncthreads();

    // ---- exact per-query top-32 within the chunk ----
    for (int q = wave; q < ROWS; q += 8) {
        float bv = -__builtin_inff();
        int   bi = -1;
        const float* row = sims[q];
        for (int g = 0; g < CHUNK / 32; ++g) {
            int   ci = c0 + g * 32 + lane;
            float v  = (ci < N) ? row[g * 32 + lane] : -__builtin_inff();
            float thr = __shfl(bv, 31);                  // current 32nd-best
            if (__any(v > thr)) {
                int idx = ci;
                bitonic_sort32_desc(v, idx, lane);
                merge32_desc(bv, bi, v, idx, lane);
            }
        }
        int qg = qt * ROWS + q;
        pV[((long)qg * nChunks + chunk) * 32 + lane] = bv;
        pI[((long)qg * nChunks + chunk) * 32 + lane] = bi;
    }
}

// Merge per-chunk partials (each group of 32 already sorted descending) into
// the final top-32 for one query. Hop-1 variant also gathers the selected
// enc_ctx rows, averages them, and emits the bf16 A matrix for hop 2.
__global__ __launch_bounds__(128)
void merge_avg_kernel(const float* __restrict__ pV, const int* __restrict__ pI,
                      int nChunks, const float* __restrict__ enc_ctx,
                      __bf16* __restrict__ newA)
{
    __shared__ int sel[KSEL];
    const int q = blockIdx.x;
    const int tid = threadIdx.x, lane = tid & 31;
    if (tid < 32) {
        float bv = -__builtin_inff(); int bi = -1;
        const float* vrow = pV + (long)q * nChunks * 32;
        const int*   irow = pI + (long)q * nChunks * 32;
        for (int g = 0; g < nChunks; ++g) {
            float v = vrow[g * 32 + lane];
            int   i = irow[g * 32 + lane];
            float thr = __shfl(bv, 31);
            if (__any(v > thr)) merge32_desc(bv, bi, v, i, lane);  // already sorted
        }
        sel[lane] = bi;
    }
    __syncthreads();
    float s = 0.f;                                        // tid = dim (0..127)
    #pragma unroll 4
    for (int j = 0; j < KSEL; ++j) s += enc_ctx[(long)sel[j] * DIME + tid];
    newA[q * DIME + tid] = (__bf16)(s * (1.0f / (float)KSEL));
}

__global__ __launch_bounds__(32)
void merge_out_kernel(const float* __restrict__ pV, const int* __restrict__ pI,
                      int nChunks, int* __restrict__ out)
{
    const int q = blockIdx.x, lane = threadIdx.x;
    float bv = -__builtin_inff(); int bi = -1;
    const float* vrow = pV + (long)q * nChunks * 32;
    const int*   irow = pI + (long)q * nChunks * 32;
    for (int g = 0; g < nChunks; ++g) {
        float v = vrow[g * 32 + lane];
        int   i = irow[g * 32 + lane];
        float thr = __shfl(bv, 31);
        if (__any(v > thr)) merge32_desc(bv, bi, v, i, lane);
    }
    out[q * KSEL + lane] = bi;                 // descending-score order, as jax top_k
}

// ------------------------------- launcher -----------------------------------

extern "C" void kernel_launch(void* const* d_in, const int* in_sizes, int n_in,
                              void* d_out, int out_size, void* d_ws, size_t ws_size,
                              hipStream_t stream) {
    const float* contexts = (const float*)d_in[0];   // [NCQ, DIME]
    const float* enc_ans  = (const float*)d_in[1];   // [N_ANS, DIME]
    const float* enc_ctx  = (const float*)d_in[2];   // [N_CTX, DIME]
    // d_in[3] = k (fixed at 32 by the reference)

    const int N_ANS = in_sizes[1] / DIME;
    const int N_CTX = in_sizes[2] / DIME;
    const int nCh1 = (N_CTX + CHUNK - 1) / CHUNK;
    const int nCh2 = (N_ANS + CHUNK - 1) / CHUNK;
    const int nChMax = nCh1 > nCh2 ? nCh1 : nCh2;

    // workspace layout
    char* ws = (char*)d_ws;
    __bf16* A1 = (__bf16*)ws;                                  // NCQ*DIME bf16
    __bf16* A2 = A1 + NCQ * DIME;                              // NCQ*DIME bf16
    float*  pV = (float*)(A2 + NCQ * DIME);                    // NCQ*nChMax*32 f32
    int*    pI = (int*)(pV + (size_t)NCQ * nChMax * 32);       // NCQ*nChMax*32 i32

    int* out = (int*)d_out;

    // hop 1
    cvt_bf16_kernel<<<(NCQ * DIME + 255) / 256, 256, 0, stream>>>(contexts, A1, NCQ * DIME);
    dim3 g1(NCQ / ROWS, nCh1);                 // query tile fastest -> L2 reuse per chunk
    score_topk_kernel<<<g1, 256, 0, stream>>>(A1, enc_ctx, pV, pI, N_CTX, nCh1);
    merge_avg_kernel<<<NCQ, 128, 0, stream>>>(pV, pI, nCh1, enc_ctx, A2);

    // hop 2
    dim3 g2(NCQ / ROWS, nCh2);
    score_topk_kernel<<<g2, 256, 0, stream>>>(A2, enc_ans, pV, pI, N_ANS, nCh2);
    merge_out_kernel<<<NCQ, 32, 0, stream>>>(pV, pI, nCh2, out);
}
